// GGNN_plus_46445776339539
// MI455X (gfx1250) — compile-verified, matrix operand order checked by reading, not run on hardware
//
#include <hip/hip_runtime.h>
#include <hip/hip_bf16.h>
#include <math.h>

// Problem constants (match reference)
#define BATCH   4
#define NNODE   4096
#define NEDGE   16384
#define DDIM    256
#define ADIM    64
#define TTYPES  16
#define HALFT   8
#define NSTEPS  3
#define ROWS    (BATCH * NNODE)   // 16384 node rows
#define EDGES   (BATCH * NEDGE)   // 65536 edges total
#define K3D     (3 * DDIM)        // 768

typedef __attribute__((ext_vector_type(16))) __bf16 v16bf;
typedef __attribute__((ext_vector_type(8)))  float  v8f;

union Frag16 { v16bf v; float4 f4[2]; };

__device__ __forceinline__ unsigned short f2bf(float x) {
  union { float f; unsigned u; } c; c.f = x;
  unsigned r = 0x7FFFu + ((c.u >> 16) & 1u);      // round-to-nearest-even
  return (unsigned short)((c.u + r) >> 16);
}
__device__ __forceinline__ float sigf(float x) { return 1.0f / (1.0f + __expf(-x)); }

// ---------------------------------------------------------------------------
// Generic f32 -> bf16 conversion
// ---------------------------------------------------------------------------
__global__ void k_cvt_bf16(const float* __restrict__ in, unsigned short* __restrict__ out, long n) {
  long i = (long)blockIdx.x * blockDim.x + threadIdx.x;
  if (i < n) out[i] = f2bf(in[i]);
}

// ---------------------------------------------------------------------------
// annotation + initial hidden state
//   ann[row, c<64]  = id>0 ? type_embed[id-1, c] : 0
//   h[row, :]       = [ann, zeros]
// ---------------------------------------------------------------------------
__global__ void k_init(const int* __restrict__ ann_id, const float* __restrict__ type_embed,
                       float* __restrict__ annf, float* __restrict__ hf) {
  long i = (long)blockIdx.x * blockDim.x + threadIdx.x;
  if (i >= (long)ROWS * DDIM) return;
  int row = (int)(i / DDIM), col = (int)(i % DDIM);
  int id = ann_id[row];
  float v = 0.0f;
  if (col < ADIM) {
    if (id > 0) v = type_embed[(id - 1) * ADIM + col];
    annf[(long)row * ADIM + col] = v;
  }
  hf[i] = (col < ADIM) ? v : 0.0f;
}

// ---------------------------------------------------------------------------
// Per-step staging: A[:,0:512]=0 ; A[:,512:768]=h ; Hbf = bf16(h)
// ---------------------------------------------------------------------------
__global__ void k_prep(const float* __restrict__ hf, float* __restrict__ Af,
                       unsigned short* __restrict__ Hbf) {
  long i = (long)blockIdx.x * blockDim.x + threadIdx.x;
  if (i >= (long)ROWS * K3D) return;
  int row = (int)(i / K3D), col = (int)(i % K3D);
  float v = 0.0f;
  if (col >= 2 * DDIM) {
    v = hf[(long)row * DDIM + (col - 2 * DDIM)];
    Hbf[(long)row * DDIM + (col - 2 * DDIM)] = f2bf(v);
  }
  Af[i] = v;
}

// ---------------------------------------------------------------------------
// WMMA bf16 GEMM:  C[M,Nout] = A[M,K] @ W[Nout,K]^T   (f32 accumulate)
// One wave computes a 16x64 tile (1 A-fragment reused for 4 WMMAs per K-step).
// ---------------------------------------------------------------------------
__global__ void k_gemm_bf16(const unsigned short* __restrict__ A,
                            const unsigned short* __restrict__ W,
                            float* __restrict__ C, int M, int Nout, int K) {
  const int wave = (int)((blockIdx.x * blockDim.x + threadIdx.x) >> 5);
  const int lane = (int)(threadIdx.x & 31);
  const int nt4  = Nout >> 6;
  const int mt   = wave / nt4;
  const int nb   = wave % nt4;
  if (mt * 16 >= M) return;                 // wave-uniform: EXEC stays all-ones
  const int half = lane >> 4;
  const int ml   = lane & 15;

  const unsigned short* arow = A + (size_t)(mt * 16 + ml) * K;
  const unsigned short* w0   = W + (size_t)(nb * 64 +  0 + ml) * K;
  const unsigned short* w1   = W + (size_t)(nb * 64 + 16 + ml) * K;
  const unsigned short* w2   = W + (size_t)(nb * 64 + 32 + ml) * K;
  const unsigned short* w3   = W + (size_t)(nb * 64 + 48 + ml) * K;

  v8f c0 = {}, c1 = {}, c2 = {}, c3 = {};
  for (int k0 = 0; k0 < K; k0 += 32) {
    const int kl = k0 + half * 8;           // lane-half K split per ISA layout
    Frag16 a, b0, b1, b2, b3;
    a.f4[0]  = *(const float4*)(arow + kl);  a.f4[1]  = *(const float4*)(arow + kl + 16);
    b0.f4[0] = *(const float4*)(w0 + kl);    b0.f4[1] = *(const float4*)(w0 + kl + 16);
    b1.f4[0] = *(const float4*)(w1 + kl);    b1.f4[1] = *(const float4*)(w1 + kl + 16);
    b2.f4[0] = *(const float4*)(w2 + kl);    b2.f4[1] = *(const float4*)(w2 + kl + 16);
    b3.f4[0] = *(const float4*)(w3 + kl);    b3.f4[1] = *(const float4*)(w3 + kl + 16);
    c0 = __builtin_amdgcn_wmma_f32_16x16x32_bf16(false, a.v, false, b0.v, (short)0, c0, false, false);
    c1 = __builtin_amdgcn_wmma_f32_16x16x32_bf16(false, a.v, false, b1.v, (short)0, c1, false, false);
    c2 = __builtin_amdgcn_wmma_f32_16x16x32_bf16(false, a.v, false, b2.v, (short)0, c2, false, false);
    c3 = __builtin_amdgcn_wmma_f32_16x16x32_bf16(false, a.v, false, b3.v, (short)0, c3, false, false);
  }
  // C/D layout: lane l -> col (l&15), rows r + 8*(l>>4)
  float* cbase = C + (size_t)(mt * 16 + half * 8) * Nout + nb * 64 + ml;
#pragma unroll
  for (int r = 0; r < 8; ++r) {
    float* crow = cbase + (size_t)r * Nout;
    crow[0] = c0[r]; crow[16] = c1[r]; crow[32] = c2[r]; crow[48] = c3[r];
  }
}

// ---------------------------------------------------------------------------
// Edge scatter for one edge-type t:
//   t <  8 : a_out[src] += Pt[dst] + bias[t]   (cols 256..511 of A)
//   t >= 8 : a_in [dst] += Pt[src] + bias[t]   (cols   0..255 of A)
// One wave per edge; 8 columns per lane; f32 global atomics.
// ---------------------------------------------------------------------------
__global__ void k_scatter(const float* __restrict__ Pt, const int* __restrict__ src,
                          const int* __restrict__ dst, const int* __restrict__ etype,
                          const float* __restrict__ edge_bias, float* __restrict__ Af, int t) {
  const int wave = (int)((blockIdx.x * blockDim.x + threadIdx.x) >> 5);
  const int lane = (int)(threadIdx.x & 31);
  if (wave >= EDGES) return;
  const int b = wave / NEDGE;
  const int e = wave - b * NEDGE;
  const int et = etype[(long)b * NEDGE + e];
  int from, to, colbase;
  if (t < HALFT) {
    if (et != t) return;
    from = dst[(long)b * NEDGE + e]; to = src[(long)b * NEDGE + e]; colbase = DDIM;
  } else {
    if (et != t - HALFT) return;
    from = src[(long)b * NEDGE + e]; to = dst[(long)b * NEDGE + e]; colbase = 0;
  }
  const float* prow = Pt + ((size_t)b * NNODE + from) * DDIM;
  const float* brow = edge_bias + (size_t)t * DDIM;
  float* arow = Af + ((size_t)b * NNODE + to) * K3D + colbase;
#pragma unroll
  for (int j = 0; j < 8; ++j) {
    int c = lane + 32 * j;
    atomicAdd(arow + c, prow[c] + brow[c]);
  }
}

// ---------------------------------------------------------------------------
// GRU gates: r=sig(Rpre+br), z=sig(Zpre+bz); stash z; write bf16(r*h) into A[:,512:768]
// ---------------------------------------------------------------------------
__global__ void k_gru_rz(const float* __restrict__ Rpre, const float* __restrict__ Zpre,
                         const float* __restrict__ br, const float* __restrict__ bz,
                         const float* __restrict__ hf, float* __restrict__ zbuf,
                         unsigned short* __restrict__ Abf) {
  long i = (long)blockIdx.x * blockDim.x + threadIdx.x;
  if (i >= (long)ROWS * DDIM) return;
  int row = (int)(i / DDIM), col = (int)(i % DDIM);
  float r = sigf(Rpre[i] + br[col]);
  float z = sigf(Zpre[i] + bz[col]);
  zbuf[i] = z;
  Abf[(long)row * K3D + 2 * DDIM + col] = f2bf(r * hf[i]);
}

// h = (1-z)*h + z*tanh(Hpre + bh)
__global__ void k_gru_update(const float* __restrict__ Hpre, const float* __restrict__ bh,
                             const float* __restrict__ zbuf, float* __restrict__ hf) {
  long i = (long)blockIdx.x * blockDim.x + threadIdx.x;
  if (i >= (long)ROWS * DDIM) return;
  int col = (int)(i % DDIM);
  float z = zbuf[i];
  float hh = tanhf(Hpre[i] + bh[col]);
  hf[i] = (1.0f - z) * hf[i] + z * hh;
}

// ---------------------------------------------------------------------------
// Readout: per-node dot([h,ann], Wa/Wo), sigmoid*tanh, sum per graph
// ---------------------------------------------------------------------------
__global__ void k_zero_acc(float* acc) { if (threadIdx.x < BATCH) acc[threadIdx.x] = 0.0f; }

__global__ void k_readout(const float* __restrict__ hf, const float* __restrict__ annf,
                          const float* __restrict__ Wa, const float* __restrict__ ba,
                          const float* __restrict__ Wo, const float* __restrict__ bo,
                          float* __restrict__ acc) {
  const int wave = (int)((blockIdx.x * blockDim.x + threadIdx.x) >> 5);
  const int lane = (int)(threadIdx.x & 31);
  if (wave >= ROWS) return;
  const int row = wave, b = row / NNODE;
  float da = 0.0f, dm = 0.0f;
#pragma unroll
  for (int j = 0; j < 8; ++j) {
    int c = lane + 32 * j;
    float v = hf[(long)row * DDIM + c];
    da += v * Wa[c]; dm += v * Wo[c];
  }
#pragma unroll
  for (int j = 0; j < 2; ++j) {
    int c = lane + 32 * j;
    float v = annf[(long)row * ADIM + c];
    da += v * Wa[DDIM + c]; dm += v * Wo[DDIM + c];
  }
  for (int off = 16; off > 0; off >>= 1) {
    da += __shfl_xor(da, off, 32);
    dm += __shfl_xor(dm, off, 32);
  }
  if (lane == 0) atomicAdd(&acc[b], sigf(da + ba[0]) * tanhf(dm + bo[0]));
}

__global__ void k_final(const float* __restrict__ acc, float* __restrict__ out) {
  if (threadIdx.x < BATCH) out[threadIdx.x] = sigf(acc[threadIdx.x]);
}

// ---------------------------------------------------------------------------
extern "C" void kernel_launch(void* const* d_in, const int* in_sizes, int n_in,
                              void* d_out, int out_size, void* d_ws, size_t ws_size,
                              hipStream_t stream) {
  const int*   ann_id     = (const int*)d_in[0];
  const int*   src        = (const int*)d_in[1];
  const int*   dst        = (const int*)d_in[2];
  const int*   etype      = (const int*)d_in[3];
  const float* edge_embed = (const float*)d_in[4];   // [16, 256*256]
  const float* edge_bias  = (const float*)d_in[5];   // [16, 256]
  const float* type_embed = (const float*)d_in[6];   // [50, 64]
  const float* Wr = (const float*)d_in[7];  const float* br = (const float*)d_in[8];
  const float* Wz = (const float*)d_in[9];  const float* bz = (const float*)d_in[10];
  const float* Wh = (const float*)d_in[11]; const float* bh = (const float*)d_in[12];
  const float* Wa = (const float*)d_in[13]; const float* ba = (const float*)d_in[14];
  const float* Wo = (const float*)d_in[15]; const float* bo = (const float*)d_in[16];
  float* out = (float*)d_out;

  // Workspace carve-out (256B aligned)
  char* ws = (char*)d_ws;
  size_t off = 0;
  auto carve = [&](size_t bytes) -> char* {
    char* p = ws + off;
    off = (off + bytes + 255) & ~(size_t)255;
    return p;
  };
  float*          hf   = (float*)carve((size_t)ROWS * DDIM * 4);
  float*          annf = (float*)carve((size_t)ROWS * ADIM * 4);
  float*          Af   = (float*)carve((size_t)ROWS * K3D * 4);
  unsigned short* Abf  = (unsigned short*)carve((size_t)ROWS * K3D * 2);
  unsigned short* Hbf  = (unsigned short*)carve((size_t)ROWS * DDIM * 2);
  float*          Pt   = (float*)carve((size_t)ROWS * DDIM * 4);
  float*          Rpre = (float*)carve((size_t)ROWS * DDIM * 4);   // also reused as Hpre
  float*          Zpre = (float*)carve((size_t)ROWS * DDIM * 4);
  float*          zbuf = (float*)carve((size_t)ROWS * DDIM * 4);
  unsigned short* WbfE = (unsigned short*)carve((size_t)TTYPES * DDIM * DDIM * 2);
  unsigned short* WbfR = (unsigned short*)carve((size_t)DDIM * K3D * 2);
  unsigned short* WbfZ = (unsigned short*)carve((size_t)DDIM * K3D * 2);
  unsigned short* WbfH = (unsigned short*)carve((size_t)DDIM * K3D * 2);
  float*          acc  = (float*)carve(64);

  const int TB = 256;
  auto blocks = [](long n, int tb) { return (int)((n + tb - 1) / tb); };

  // One-time weight conversions (deterministic every call)
  k_cvt_bf16<<<blocks((long)TTYPES * DDIM * DDIM, TB), TB, 0, stream>>>(edge_embed, WbfE, (long)TTYPES * DDIM * DDIM);
  k_cvt_bf16<<<blocks((long)DDIM * K3D, TB), TB, 0, stream>>>(Wr, WbfR, (long)DDIM * K3D);
  k_cvt_bf16<<<blocks((long)DDIM * K3D, TB), TB, 0, stream>>>(Wz, WbfZ, (long)DDIM * K3D);
  k_cvt_bf16<<<blocks((long)DDIM * K3D, TB), TB, 0, stream>>>(Wh, WbfH, (long)DDIM * K3D);

  k_init<<<blocks((long)ROWS * DDIM, TB), TB, 0, stream>>>(ann_id, type_embed, annf, hf);

  const int gemm_waves_proj = (ROWS / 16) * (DDIM / 64);   // 4096 waves
  const int gemm_blocks     = gemm_waves_proj * 32 / TB;   // 512 blocks
  const int scat_blocks     = EDGES * 32 / TB;             // 8192 blocks

  for (int s = 0; s < NSTEPS; ++s) {
    k_prep<<<blocks((long)ROWS * K3D, TB), TB, 0, stream>>>(hf, Af, Hbf);

    for (int t = 0; t < TTYPES; ++t) {
      k_gemm_bf16<<<gemm_blocks, TB, 0, stream>>>(Hbf, WbfE + (size_t)t * DDIM * DDIM, Pt,
                                                  ROWS, DDIM, DDIM);
      k_scatter<<<scat_blocks, TB, 0, stream>>>(Pt, src, dst, etype, edge_bias, Af, t);
    }

    k_cvt_bf16<<<blocks((long)ROWS * K3D, TB), TB, 0, stream>>>(Af, Abf, (long)ROWS * K3D);

    k_gemm_bf16<<<gemm_blocks, TB, 0, stream>>>(Abf, WbfR, Rpre, ROWS, DDIM, K3D);
    k_gemm_bf16<<<gemm_blocks, TB, 0, stream>>>(Abf, WbfZ, Zpre, ROWS, DDIM, K3D);
    k_gru_rz<<<blocks((long)ROWS * DDIM, TB), TB, 0, stream>>>(Rpre, Zpre, br, bz, hf, zbuf, Abf);
    k_gemm_bf16<<<gemm_blocks, TB, 0, stream>>>(Abf, WbfH, Rpre /*Hpre*/, ROWS, DDIM, K3D);
    k_gru_update<<<blocks((long)ROWS * DDIM, TB), TB, 0, stream>>>(Rpre, bh, zbuf, hf);
  }

  k_zero_acc<<<1, 32, 0, stream>>>(acc);
  k_readout<<<ROWS * 32 / TB, TB, 0, stream>>>(hf, annf, Wa, ba, Wo, bo, acc);
  k_final<<<1, 32, 0, stream>>>(acc, out);
}
